// Net_17978733101342
// MI455X (gfx1250) — matrix operand order repaired and checked by reading, not compiled, not run on hardware
//
#include <hip/hip_runtime.h>
#include <hip/hip_bf16.h>

typedef __attribute__((ext_vector_type(16))) _Float16 v16h;
typedef __attribute__((ext_vector_type(8)))  float    v8f;

#define BATCH 2
#define CH    64
#define CH4   256
#define IH    256
#define IW    256
#define HWSZ  ((size_t)IH * (size_t)IW)
#define BN_EPS 1e-5f

// 32-byte WMMA 16-bit operand fragment, loadable as two 16B vectors.
union AF {
  uint4 u[2];
  v16h  v;
};

// ---------- WMMA fragment layout (CDNA5 ISA §7.12.2) ----------
// 16-bit A (16x32): lane m = lane&15; halves = two contiguous 8-K runs:
//   run0 K = (lane>=16 ? 8 : 0) + [0..7],  run1 = run0 + 16.
// 16-bit B (32x16): lane n = lane&15; same K order as A; this code loads 16 contiguous K at (lane>=16 ? 16 : 0) for both.
// f32 C/D (16x16): reg r -> M = (lane>=16 ? 8 : 0) + r, N = lane&15.
__device__ __forceinline__ int d_row(int lane, int r) { return ((lane >> 4) << 3) + r; }

__device__ __forceinline__ v8f wmma16(v16h a, v16h b, v8f c) {
  return __builtin_amdgcn_wmma_f32_16x16x32_f16(false, a, false, b, (short)0, c, false, false);
}

// Load A fragment: base points at K=0 of a 32-K contiguous f16 run (16B aligned).
__device__ __forceinline__ v16h load_a(const _Float16* base, int hi) {
  AF a;
  a.u[0] = *(const uint4*)(base + hi * 16);
  a.u[1] = *(const uint4*)(base + hi * 16 + 8);
  return a.v;
}
// Load B fragment: base points at K=0 of a 32-K contiguous f16 run (16B aligned).
__device__ __forceinline__ v16h load_b(const _Float16* base, int hi) {
  AF b;
  b.u[0] = *(const uint4*)(base + hi * 16);
  b.u[1] = *(const uint4*)(base + hi * 16 + 8);
  return b.v;
}

// ---------------------------------------------------------------------------
// Weight pre-pack: (256,64,3,3) f32, k=cin*9+tap  ->  f16 wp[o][tap*64+cin]
// so conv3x3 B-fragments are two 16B loads. Grid: 576 x 256 threads (exact).
// ---------------------------------------------------------------------------
__global__ void __launch_bounds__(256) pack_weights(
    const float* __restrict__ w, _Float16* __restrict__ wp)
{
  const int i = blockIdx.x * 256 + threadIdx.x;    // over 256*576
  const int o = i / 576, k = i - o * 576;          // k = new order
  const int tap = k >> 6, cin = k & 63;
  wp[(size_t)o * 576 + k] = (_Float16)w[(size_t)o * 576 + cin * 9 + tap];
}

// ---------------------------------------------------------------------------
// Grouped 3x3 conv (groups=4, 64 in / 64 out per group), implicit GEMM.
// mode 0: input = BN(catfea) (f32 src), LeakyReLU epilogue.
// mode 1: input = t1 (f16), epilogue adds BN(catfea) residual.
// Block: 8 waves = 2 pixel-tiles x 4 outch-tiles -> 32 px x 64 outch.
// Grid: (H * W/32, groups, batch). K-order: k = tap*64 + cin.
// ---------------------------------------------------------------------------
__global__ void __launch_bounds__(256) conv3x3_grouped(
    const float* __restrict__ cat,      // (B,256,H,W) catfea (BN source)
    const _Float16* __restrict__ in16,  // (B,256,H,W) f16 input (mode 1)
    const _Float16* __restrict__ wpk,   // packed weights (256,576) f16
    const float* __restrict__ bias,     // (256)
    const float* __restrict__ bn_g, const float* __restrict__ bn_b,
    const float* __restrict__ bn_m, const float* __restrict__ bn_v,
    _Float16* __restrict__ out16,       // (B,256,H,W) f16
    int mode)
{
  const int tid = threadIdx.x, lane = tid & 31, wv = tid >> 5;
  const int g  = blockIdx.y, bz = blockIdx.z;
  const int h0 = blockIdx.x >> 3;            // IW/32 == 8 strips per row
  const int w0 = (blockIdx.x & 7) << 5;

  // [ky][col(-1..32)][cin], cin innermost, pitch 72 halves (16B aligned rows)
  __shared__ __align__(16) _Float16 tile[3][40][72];

  // ---- stage input tile (halo, zero-padded); coalesced global reads ----
  for (int e = tid; e < 64 * 3 * 34; e += 256) {
    const int c  = e / 102;
    const int rm = e - c * 102;
    const int ry = rm / 34, cx = rm - ry * 34;
    const int gh = h0 - 1 + ry, gw = w0 - 1 + cx;
    float f = 0.0f;
    if (gh >= 0 && gh < IH && gw >= 0 && gw < IW) {
      const int ch = g * 64 + c;
      if (mode == 0) {
        const float x  = cat[(((size_t)bz * CH4 + ch) * IH + gh) * IW + gw];
        const float sc = bn_g[ch] * rsqrtf(bn_v[ch] + BN_EPS);
        f = (x - bn_m[ch]) * sc + bn_b[ch];
      } else {
        f = (float)in16[(((size_t)bz * CH4 + ch) * IH + gh) * IW + gw];
      }
    }
    tile[ry][cx][c] = (_Float16)f;
  }
  __syncthreads();

  const int pt = wv & 1;                 // pixel tile (16 cols)
  const int ot = wv >> 1;                // outch tile
  const int obase = g * 64 + ot * 16;
  const int n  = lane & 15;
  const int hi = lane >> 4;
  const int col0 = pt * 16 + (lane & 15);  // lane's M -> tile col (before +kx)

  const _Float16* wrow = wpk + (size_t)(obase + n) * 576;

  v8f acc = {0.f,0.f,0.f,0.f,0.f,0.f,0.f,0.f};
#pragma unroll
  for (int tap = 0; tap < 9; ++tap) {
    const int ky = tap / 3, kx = tap - ky * 3;
    const _Float16* arow = &tile[ky][col0 + kx][0];
    const _Float16* brow = wrow + tap * 64;
#pragma unroll
    for (int kc = 0; kc < 64; kc += 32) {
      acc = wmma16(load_a(arow + kc, hi), load_b(brow + kc, hi), acc);
    }
  }

  // ---- epilogue ----
  const int och = obase + n;
  const float bv = bias[och];
  float rsc = 0.f, rsh = 0.f;
  if (mode == 1) {
    rsc = bn_g[och] * rsqrtf(bn_v[och] + BN_EPS);
    rsh = bn_b[och] - bn_m[och] * rsc;
  }
#pragma unroll
  for (int r = 0; r < 8; ++r) {
    const int colw = w0 + pt * 16 + d_row(lane, r);
    float v = acc[r] + bv;
    if (mode == 0) {
      v = (v >= 0.f) ? v : 0.1f * v;
    } else {
      const float x = cat[(((size_t)bz * CH4 + och) * IH + h0) * IW + colw];
      v += x * rsc + rsh;  // + BN(catfea) residual
    }
    out16[(((size_t)bz * CH4 + och) * IH + h0) * IW + colw] = (_Float16)v;
  }
}

// ---------------------------------------------------------------------------
// Grouped 1x1 conv: groups=4, 64 in -> 16 out per group. WMMA GEMM.
// Block: 8 waves x (16 px x 16 outch) = 128 px. Grid: (HW/128, 4, B)
// ---------------------------------------------------------------------------
__global__ void __launch_bounds__(256) conv1x1_grouped(
    const _Float16* __restrict__ in16,  // (B,256,HW)
    const float* __restrict__ wgt,      // (64,64)
    const float* __restrict__ bias,     // (64)
    float* __restrict__ outf)           // (B,64,HW)
{
  const int tid = threadIdx.x, lane = tid & 31, wv = tid >> 5;
  const int g = blockIdx.y, bz = blockIdx.z;
  const int pbase0 = blockIdx.x * 128;
  const int n  = lane & 15;
  const int hi = lane >> 4;

  __shared__ __align__(16) _Float16 alds[128][72];  // [px][cin]

  for (int e = tid; e < 128 * 64; e += 256) {
    const int px = e & 127, cin = e >> 7;   // coalesced over px per channel
    alds[px][cin] =
        in16[((size_t)bz * CH4 + g * 64 + cin) * HWSZ + pbase0 + px];
  }
  __syncthreads();

  const _Float16* arow = &alds[wv * 16 + (lane & 15)][0];
  v8f acc = {0.f,0.f,0.f,0.f,0.f,0.f,0.f,0.f};
#pragma unroll
  for (int kc = 0; kc < 64; kc += 32) {
    v16h bf;
    const float* wp = wgt + (size_t)(g * 16 + n) * 64 + kc + hi * 16;
#pragma unroll
    for (int e16 = 0; e16 < 16; ++e16) bf[e16] = (_Float16)wp[e16];
    acc = wmma16(load_a(arow + kc, hi), bf, acc);
  }

  const int och = g * 16 + n;
  const float bv = bias[och];
#pragma unroll
  for (int r = 0; r < 8; ++r) {
    const int p = pbase0 + wv * 16 + d_row(lane, r);
    outf[((size_t)bz * CH + och) * HWSZ + p] = acc[r] + bv;
  }
}

// ---------------------------------------------------------------------------
// Fused per-patch parallax attention. One block (8 waves) per 8x8 patch.
// 4 WMMA matmuls (64x64x64), softmax, m_relax + V, tanh, warp, blend.
// Gathered operands staged once in a reused LDS buffer -> vector B-frag loads.
// Grid: (B * H/8 * W/8)
// ---------------------------------------------------------------------------
__global__ void __launch_bounds__(256) pam_attention(
    const float* __restrict__ Q, const float* __restrict__ K,
    const float* __restrict__ xl, const float* __restrict__ xr,
    const float* __restrict__ dl, const float* __restrict__ dr,
    float* __restrict__ out)
{
  const int tid = threadIdx.x, lane = tid & 31, wv = tid >> 5;
  int pbk = blockIdx.x;
  const int wb = pbk & 31; pbk >>= 5;       // W/8 == 32
  const int hb = pbk & 31; pbk >>= 5;       // H/8 == 32
  const int bz = pbk;
  const int hi = lane >> 4;

  __shared__ float sc[64][65];                          // f32 stage / scores
  __shared__ __align__(16) _Float16 qc[64][72];         // centered Qp
  __shared__ __align__(16) _Float16 kc[64][72];         // centered Kp
  __shared__ __align__(16) _Float16 m1[64][72];         // Mr2l
  __shared__ __align__(16) _Float16 m2[64][72];         // Ml2r
  __shared__ __align__(16) _Float16 sel[64][72];        // reused gather buffer
  __shared__ int   idxr[64], idxl[64];
  __shared__ float Vl[64], Vr[64], mean8[8];

#define HOFP(p) (hb * 8 + ((p) >> 3))
#define WOFP(p) (wb * 8 + ((p) & 7))

  // ---- disparity gather indices ----
  if (tid < 64) {
    const int h = HOFP(tid), w = WOFP(tid);
    const float dlv = dl[((size_t)bz * IH + h) * IW + w];
    const float drv = dr[((size_t)bz * IH + h) * IW + w];
    idxr[tid] = (int)fmaxf((float)w + 0.5f - dlv, 0.0f);
    idxl[tid] = (int)fminf((float)w + 0.5f + drv, (float)(IW - 1));
  }
  __syncthreads();

  // ---- stage Qp, mean over (ws_w, c) per ws-row, center, f16 ----
  for (int e = tid; e < 4096; e += 256) {
    const int p = e >> 6, c = e & 63;
    sc[p][c] = Q[((size_t)bz * CH + c) * HWSZ + (size_t)HOFP(p) * IW + WOFP(p)];
  }
  __syncthreads();
  if (tid < 8) {
    float s = 0.f;
    for (int pp = 0; pp < 8; ++pp)
      for (int c = 0; c < 64; ++c) s += sc[tid * 8 + pp][c];
    mean8[tid] = s * (1.0f / 512.0f);
  }
  __syncthreads();
  for (int e = tid; e < 4096; e += 256) {
    const int p = e >> 6, c = e & 63;
    qc[p][c] = (_Float16)(sc[p][c] - mean8[p >> 3]);
  }
  __syncthreads();

  // ---- stage Kp likewise ----
  for (int e = tid; e < 4096; e += 256) {
    const int p = e >> 6, c = e & 63;
    sc[p][c] = K[((size_t)bz * CH + c) * HWSZ + (size_t)HOFP(p) * IW + WOFP(p)];
  }
  __syncthreads();
  if (tid < 8) {
    float s = 0.f;
    for (int pp = 0; pp < 8; ++pp)
      for (int c = 0; c < 64; ++c) s += sc[tid * 8 + pp][c];
    mean8[tid] = s * (1.0f / 512.0f);
  }
  __syncthreads();
  for (int e = tid; e < 4096; e += 256) {
    const int p = e >> 6, c = e & 63;
    kc[p][c] = (_Float16)(sc[p][c] - mean8[p >> 3]);
  }
  __syncthreads();

  // ---- stage Ksel: sel[q][c] = K[b, c, h(q), idxr[q]] ----
  for (int e = tid; e < 4096; e += 256) {
    const int c = e & 63, q = e >> 6;
    sel[q][c] = (_Float16)K[((size_t)bz * CH + c) * HWSZ +
                            (size_t)HOFP(q) * IW + idxr[q]];
  }
  __syncthreads();

  // ---- score_r2l = Qp_centered @ Ksel^T ----
  for (int t = wv; t < 16; t += 8) {
    const int Mt = t >> 2, Nt = t & 3;
    const _Float16* arow = &qc[Mt * 16 + (lane & 15)][0];
    const _Float16* brow = &sel[Nt * 16 + (lane & 15)][0];   // B[k=c][n=q]
    v8f acc = {0.f,0.f,0.f,0.f,0.f,0.f,0.f,0.f};
#pragma unroll
    for (int kc2 = 0; kc2 < 64; kc2 += 32)
      acc = wmma16(load_a(arow + kc2, hi), load_b(brow + kc2, hi), acc);
#pragma unroll
    for (int r = 0; r < 8; ++r)
      sc[Mt * 16 + d_row(lane, r)][Nt * 16 + (lane & 15)] = acc[r];
  }
  __syncthreads();

  // ---- softmax rows -> Mr2l (f16) ----
  if (tid < 64) {
    float mx = -3.0e38f;
    for (int k = 0; k < 64; ++k) mx = fmaxf(mx, sc[tid][k]);
    float s = 0.f;
    for (int k = 0; k < 64; ++k) { const float e = __expf(sc[tid][k] - mx); sc[tid][k] = e; s += e; }
    const float inv = 1.0f / s;
    for (int k = 0; k < 64; ++k) m1[tid][k] = (_Float16)(sc[tid][k] * inv);
  }
  __syncthreads();

  // ---- stage Qsel: sel[q][c] = Q[b, c, h(q), idxl[q]] ----
  for (int e = tid; e < 4096; e += 256) {
    const int c = e & 63, q = e >> 6;
    sel[q][c] = (_Float16)Q[((size_t)bz * CH + c) * HWSZ +
                            (size_t)HOFP(q) * IW + idxl[q]];
  }
  __syncthreads();

  // ---- score_l2r = Kp_centered @ Qsel^T ----
  for (int t = wv; t < 16; t += 8) {
    const int Mt = t >> 2, Nt = t & 3;
    const _Float16* arow = &kc[Mt * 16 + (lane & 15)][0];
    const _Float16* brow = &sel[Nt * 16 + (lane & 15)][0];
    v8f acc = {0.f,0.f,0.f,0.f,0.f,0.f,0.f,0.f};
#pragma unroll
    for (int kc2 = 0; kc2 < 64; kc2 += 32)
      acc = wmma16(load_a(arow + kc2, hi), load_b(brow + kc2, hi), acc);
#pragma unroll
    for (int r = 0; r < 8; ++r)
      sc[Mt * 16 + d_row(lane, r)][Nt * 16 + (lane & 15)] = acc[r];
  }
  __syncthreads();
  if (tid < 64) {
    float mx = -3.0e38f;
    for (int k = 0; k < 64; ++k) mx = fmaxf(mx, sc[tid][k]);
    float s = 0.f;
    for (int k = 0; k < 64; ++k) { const float e = __expf(sc[tid][k] - mx); sc[tid][k] = e; s += e; }
    const float inv = 1.0f / s;
    for (int k = 0; k < 64; ++k) m2[tid][k] = (_Float16)(sc[tid][k] * inv);
  }
  __syncthreads();

  // ---- V_left/V_right: m_relax (+-2 rows) einsum, tanh(5*) ----
  if (tid < 64) {
    const int i = tid;
    float vl = 0.f, vr = 0.f;
    for (int k = 0; k < 64; ++k) {
      float s1 = 0.f, s2 = 0.f;
#pragma unroll
      for (int o = -2; o <= 2; ++o) {
        const int j = i + o;
        if (j >= 0 && j < 64) { s1 += (float)m1[j][k]; s2 += (float)m2[j][k]; }
      }
      vl += s1 * (float)m2[k][i];
      vr += s2 * (float)m1[k][i];
    }
    Vl[i] = tanhf(5.0f * vl);
    Vr[i] = tanhf(5.0f * vr);
  }
  __syncthreads();

  float* outL = out;
  float* outR = out + (size_t)BATCH * CH * HWSZ;

  // ---- stage xr_sel transposed: sel[c][q] = xr[b, c, h(q), idxr[q]] ----
  for (int e = tid; e < 4096; e += 256) {
    const int q = e & 63, c = e >> 6;
    sel[c][q] = (_Float16)xr[((size_t)bz * CH + c) * HWSZ +
                             (size_t)HOFP(q) * IW + idxr[q]];
  }
  __syncthreads();

  // ---- x_leftT = Mr2l @ xr_sel ; blend -> out_left ----
  for (int t = wv; t < 16; t += 8) {
    const int Mt = t >> 2, Nt = t & 3;
    const int c = Nt * 16 + (lane & 15);
    const _Float16* arow = &m1[Mt * 16 + (lane & 15)][0];
    const _Float16* brow = &sel[c][0];                 // B[k=q][n=c]
    v8f acc = {0.f,0.f,0.f,0.f,0.f,0.f,0.f,0.f};
#pragma unroll
    for (int kc2 = 0; kc2 < 64; kc2 += 32)
      acc = wmma16(load_a(arow + kc2, hi), load_b(brow + kc2, hi), acc);
#pragma unroll
    for (int r = 0; r < 8; ++r) {
      const int p = Mt * 16 + d_row(lane, r);
      const float v = Vl[p];
      const size_t gi = ((size_t)bz * CH + c) * HWSZ + (size_t)HOFP(p) * IW + WOFP(p);
      outL[gi] = xl[gi] * (1.0f - v) + acc[r] * v;
    }
  }
  __syncthreads();

  // ---- stage xl_sel transposed: sel[c][q] = xl[b, c, h(q), idxl[q]] ----
  for (int e = tid; e < 4096; e += 256) {
    const int q = e & 63, c = e >> 6;
    sel[c][q] = (_Float16)xl[((size_t)bz * CH + c) * HWSZ +
                             (size_t)HOFP(q) * IW + idxl[q]];
  }
  __syncthreads();

  // ---- x_rightT = Ml2r @ xl_sel ; blend -> out_right ----
  for (int t = wv; t < 16; t += 8) {
    const int Mt = t >> 2, Nt = t & 3;
    const int c = Nt * 16 + (lane & 15);
    const _Float16* arow = &m2[Mt * 16 + (lane & 15)][0];
    const _Float16* brow = &sel[c][0];
    v8f acc = {0.f,0.f,0.f,0.f,0.f,0.f,0.f,0.f};
#pragma unroll
    for (int kc2 = 0; kc2 < 64; kc2 += 32)
      acc = wmma16(load_a(arow + kc2, hi), load_b(brow + kc2, hi), acc);
#pragma unroll
    for (int r = 0; r < 8; ++r) {
      const int p = Mt * 16 + d_row(lane, r);
      const float v = Vr[p];
      const size_t gi = ((size_t)bz * CH + c) * HWSZ + (size_t)HOFP(p) * IW + WOFP(p);
      outR[gi] = xr[gi] * (1.0f - v) + acc[r] * v;
    }
  }
#undef HOFP
#undef WOFP
}

// ---------------------------------------------------------------------------
extern "C" void kernel_launch(void* const* d_in, const int* in_sizes, int n_in,
                              void* d_out, int out_size, void* d_ws, size_t ws_size,
                              hipStream_t stream) {
  (void)in_sizes; (void)n_in; (void)out_size; (void)ws_size;

  const float* x_left  = (const float*)d_in[0];
  const float* x_right = (const float*)d_in[1];
  const float* cat_l   = (const float*)d_in[2];
  const float* cat_r   = (const float*)d_in[3];
  const float* d_left  = (const float*)d_in[4];
  const float* d_right = (const float*)d_in[5];
  const float* bn_g    = (const float*)d_in[6];
  const float* bn_b    = (const float*)d_in[7];
  const float* bn_m    = (const float*)d_in[8];
  const float* bn_v    = (const float*)d_in[9];
  const float* rb_w1   = (const float*)d_in[10];
  const float* rb_b1   = (const float*)d_in[11];
  const float* rb_w2   = (const float*)d_in[12];
  const float* rb_b2   = (const float*)d_in[13];
  const float* bq_w    = (const float*)d_in[14];
  const float* bq_b    = (const float*)d_in[15];
  const float* bs_w    = (const float*)d_in[16];
  const float* bs_b    = (const float*)d_in[17];
  float* out = (float*)d_out;

  char* ws = (char*)d_ws;
  const size_t t_bytes = (size_t)BATCH * CH4 * HWSZ * sizeof(_Float16); // 64 MiB
  const size_t q_bytes = (size_t)BATCH * CH  * HWSZ * sizeof(float);    // 32 MiB
  const size_t w_bytes = (size_t)CH4 * 576 * sizeof(_Float16);          // 288 KiB
  _Float16* t1  = (_Float16*)ws;
  _Float16* t2  = (_Float16*)(ws + t_bytes);
  float*    Qb  = (float*)(ws + 2 * t_bytes);
  float*    Kb  = (float*)(ws + 2 * t_bytes + q_bytes);
  _Float16* wp1 = (_Float16*)(ws + 2 * t_bytes + 2 * q_bytes);
  _Float16* wp2 = (_Float16*)(ws + 2 * t_bytes + 2 * q_bytes + w_bytes);

  const dim3 cgrid(IH * (IW / 32), 4, BATCH);
  const dim3 ggrid((unsigned)(HWSZ / 128), 4, BATCH);
  const dim3 agrid(BATCH * (IH / 8) * (IW / 8));

  // Pack conv3x3 weights to f16 fragment order
  pack_weights<<<dim3(576), 256, 0, stream>>>(rb_w1, wp1);
  pack_weights<<<dim3(576), 256, 0, stream>>>(rb_w2, wp2);

  // Left branch: BN -> resb -> Q
  conv3x3_grouped<<<cgrid, 256, 0, stream>>>(cat_l, t1, wp1, rb_b1, bn_g, bn_b, bn_m, bn_v, t1, 0);
  conv3x3_grouped<<<cgrid, 256, 0, stream>>>(cat_l, t1, wp2, rb_b2, bn_g, bn_b, bn_m, bn_v, t2, 1);
  conv1x1_grouped<<<ggrid, 256, 0, stream>>>(t2, bq_w, bq_b, Qb);

  // Right branch: BN -> resb -> K
  conv3x3_grouped<<<cgrid, 256, 0, stream>>>(cat_r, t1, wp1, rb_b1, bn_g, bn_b, bn_m, bn_v, t1, 0);
  conv3x3_grouped<<<cgrid, 256, 0, stream>>>(cat_r, t1, wp2, rb_b2, bn_g, bn_b, bn_m, bn_v, t2, 1);
  conv1x1_grouped<<<ggrid, 256, 0, stream>>>(t2, bs_w, bs_b, Kb);

  // Fused per-patch attention + warping + blend
  pam_attention<<<agrid, 256, 0, stream>>>(Qb, Kb, x_left, x_right, d_left, d_right, out);
}
